// GINNet_47957604827354
// MI455X (gfx1250) — matrix operand order repaired and checked by reading, not compiled, not run on hardware
//
#include <hip/hip_runtime.h>
#include <hip/hip_bf16.h>

typedef __attribute__((ext_vector_type(16))) __bf16 v16bf;
typedef __attribute__((ext_vector_type(8)))  float  v8f;

#define DFEAT 128
#define NGRAPH 64

// ---------------------------------------------------------------- utilities
__global__ void copy_f4(const float4* __restrict__ in, float4* __restrict__ out, int n4) {
    int i = blockIdx.x * blockDim.x + threadIdx.x;
    int stride = gridDim.x * blockDim.x;
    for (; i < n4; i += stride) out[i] = in[i];
}

__global__ void zero_f(float* __restrict__ p, int n) {
    int i = blockIdx.x * 256 + threadIdx.x;
    if (i < n) p[i] = 0.0f;
}

// Build the bf16 B-fragment-swizzled copy of one 128x128 weight matrix.
// Fragment f = (tile n in 0..7, kb in 0..3, lane in 0..31): 16 contiguous bf16
// at Wf[((n*4+kb)*32+lane)*16 + e], where half = lane>>4, lr = lane&15,
// col = n*16+lr, K = kb*32 + (e<8 ? half*8+e : 16+half*8+(e-8)).
__global__ void build_wfrag(const float* __restrict__ W, __bf16* __restrict__ Wf) {
    int i = blockIdx.x * 256 + threadIdx.x;          // 0 .. 16383
    if (i >= DFEAT * DFEAT) return;
    int e    = i & 15;
    int lane = (i >> 4) & 31;
    int kb   = (i >> 9) & 3;
    int n    = i >> 11;
    int half = lane >> 4, lr = lane & 15;
    int col  = n * 16 + lr;
    int k    = kb * 32 + (e < 8 ? half * 8 + e : 16 + half * 8 + (e - 8));
    Wf[i] = (__bf16)W[k * DFEAT + col];
}

// ------------------------------------------------- edge scatter-add (HBM bound)
__global__ __launch_bounds__(256) void scatter_add_edges(
        const float* __restrict__ h,
        const long long* __restrict__ src,
        const long long* __restrict__ dst,
        float* __restrict__ agg, int E) {
    int edge = blockIdx.x * 8 + (threadIdx.x >> 5);
    if (edge >= E) return;
    int lane = threadIdx.x & 31;
    long long s = src[edge];
    long long d = dst[edge];
    float4 v = ((const float4*)(h + s * (long long)DFEAT))[lane];
    float* o = agg + d * (long long)DFEAT + lane * 4;
    atomicAdd(o + 0, v.x);
    atomicAdd(o + 1, v.y);
    atomicAdd(o + 2, v.z);
    atomicAdd(o + 3, v.w);
}

// ------------------------------------------------- WMMA GEMM: out = relu(A@W + b)
// Block: 256 thr = 8 waves, owns 32 rows x 128 cols. Wave w owns cols
// [16w,16w+16) for both 16-row halves, sharing each B fragment across 2 WMMAs.
template <typename TIN, typename TOUT>
__global__ __launch_bounds__(256) void gemm_bias_relu(
        const TIN* __restrict__ A,
        const __bf16* __restrict__ Wf,
        const float* __restrict__ bias,
        TOUT* __restrict__ out, int N) {
    __shared__ __bf16 As[32][DFEAT + 8];   // pitch 272B = 17*16B: aligned + conflict-free

    int rowBase = blockIdx.x * 32;

    if constexpr (sizeof(TIN) == 2) {
        // bf16 input: pure copy -> CDNA5 async global->LDS DMA (ASYNCcnt).
        // 32 rows x 16 chunks of 16B; 512 chunks over 256 threads.
        for (int i = threadIdx.x; i < 32 * (DFEAT / 8); i += 256) {
            int r  = i >> 4;
            int c8 = i & 15;
            unsigned ldsoff = (unsigned)(uintptr_t)&As[r][c8 * 8];   // low 32b = LDS offset
            const TIN* gp = A + (long long)(rowBase + r) * DFEAT + c8 * 8;
            asm volatile("global_load_async_to_lds_b128 %0, %1, off"
                         :: "v"(ldsoff), "v"(gp) : "memory");
        }
        asm volatile("s_wait_asynccnt 0" ::: "memory");
    } else {
        // fp32 input: coalesced load + convert to bf16.
        for (int i = threadIdx.x; i < 32 * DFEAT; i += 256) {
            int r = i >> 7, c = i & (DFEAT - 1);
            int gr = rowBase + r;
            float v = (gr < N) ? (float)A[(long long)gr * DFEAT + c] : 0.0f;
            As[r][c] = (__bf16)v;
        }
    }
    __syncthreads();

    int wave = threadIdx.x >> 5;       // col tile 0..7
    int lane = threadIdx.x & 31;
    int half = lane >> 4;
    int lr   = lane & 15;
    int col  = wave * 16 + lr;

    const uint4* rp0 = (const uint4*)&As[lr][0];        // rows 0..15
    const uint4* rp1 = (const uint4*)&As[lr + 16][0];   // rows 16..31
    const uint4* wq  = (const uint4*)Wf;

    v8f acc0 = {}, acc1 = {};
#pragma unroll
    for (int kb = 0; kb < 4; ++kb) {
        union { v16bf v; uint4 q[2]; } ub;              // B: 32B contiguous per lane
        int fidx = ((wave * 4 + kb) * 32 + lane) * 2;
        ub.q[0] = wq[fidx];
        ub.q[1] = wq[fidx + 1];

        int q0 = (kb * 32 + half * 8) >> 3;             // A: two 16B LDS loads each
        union { v16bf v; uint4 q[2]; } ua0, ua1;
        ua0.q[0] = rp0[q0];     ua0.q[1] = rp0[q0 + 2];
        ua1.q[0] = rp1[q0];     ua1.q[1] = rp1[q0 + 2];

        acc0 = __builtin_amdgcn_wmma_f32_16x16x32_bf16(false, ua0.v, false, ub.v,
                                                       (short)0, acc0, false, false);
        acc1 = __builtin_amdgcn_wmma_f32_16x16x32_bf16(false, ua1.v, false, ub.v,
                                                       (short)0, acc1, false, false);
    }

    float bcol = bias[col];
    if (rowBase + 32 <= N) {
        // Bulk path: no per-row guards (only the last of ~1563 blocks is partial).
#pragma unroll
        for (int i = 0; i < 8; ++i) {
            int r0 = rowBase + half * 8 + i;
            out[(long long)r0 * DFEAT + col]        = (TOUT)fmaxf(acc0[i] + bcol, 0.0f);
            out[(long long)(r0 + 16) * DFEAT + col] = (TOUT)fmaxf(acc1[i] + bcol, 0.0f);
        }
    } else {
#pragma unroll
        for (int i = 0; i < 8; ++i) {
            int r0 = rowBase + half * 8 + i;
            int r1 = r0 + 16;
            if (r0 < N) out[(long long)r0 * DFEAT + col] = (TOUT)fmaxf(acc0[i] + bcol, 0.0f);
            if (r1 < N) out[(long long)r1 * DFEAT + col] = (TOUT)fmaxf(acc1[i] + bcol, 0.0f);
        }
    }
}

// ------------------------------------------------- mean-pool scatter
__global__ __launch_bounds__(256) void pool_scatter(
        const float* __restrict__ h, const long long* __restrict__ batch,
        float* __restrict__ sums, float* __restrict__ counts, int N) {
    int node = blockIdx.x * 2 + (threadIdx.x >> 7);
    if (node >= N) return;
    int f = threadIdx.x & (DFEAT - 1);
    long long g = batch[node];
    atomicAdd(&sums[g * DFEAT + f], h[(long long)node * DFEAT + f]);
    if (f == 0) atomicAdd(&counts[g], 1.0f);
}

// ------------------------------------------------- FFN + softmax (tiny: 64 rows)
__global__ __launch_bounds__(128) void ffn_softmax(
        const float* __restrict__ sums, const float* __restrict__ counts,
        const float* __restrict__ wf1, const float* __restrict__ bf1,
        const float* __restrict__ wf2, const float* __restrict__ bf2,
        float* __restrict__ out) {
    __shared__ float p[DFEAT];
    __shared__ float t[DFEAT];
    __shared__ float logits[16];
    int g = blockIdx.x;
    int tid = threadIdx.x;

    float cnt = fmaxf(counts[g], 1.0f);
    p[tid] = sums[g * DFEAT + tid] / cnt;
    __syncthreads();

    float acc = bf1[tid];
    for (int k = 0; k < DFEAT; ++k) acc += p[k] * wf1[k * DFEAT + tid];
    t[tid] = fmaxf(acc, 0.0f);
    __syncthreads();

    if (tid < 10) {
        float a = bf2[tid];
        for (int k = 0; k < DFEAT; ++k) a += t[k] * wf2[k * 10 + tid];
        logits[tid] = a;
    }
    __syncthreads();

    if (tid == 0) {
        float m = logits[0];
        for (int i = 1; i < 10; ++i) m = fmaxf(m, logits[i]);
        float s = 0.0f, e[10];
        for (int i = 0; i < 10; ++i) { e[i] = expf(logits[i] - m); s += e[i]; }
        float inv = 1.0f / s;
        for (int i = 0; i < 10; ++i) out[g * 10 + i] = e[i] * inv;
    }
}

// ---------------------------------------------------------------- launcher
extern "C" void kernel_launch(void* const* d_in, const int* in_sizes, int n_in,
                              void* d_out, int out_size, void* d_ws, size_t ws_size,
                              hipStream_t stream) {
    const float*     x     = (const float*)d_in[0];
    const long long* ei    = (const long long*)d_in[1];  // int64 per reference
    const long long* batch = (const long long*)d_in[2];
    const float* wa[3] = {(const float*)d_in[3],  (const float*)d_in[7],  (const float*)d_in[11]};
    const float* ba[3] = {(const float*)d_in[4],  (const float*)d_in[8],  (const float*)d_in[12]};
    const float* wb[3] = {(const float*)d_in[5],  (const float*)d_in[9],  (const float*)d_in[13]};
    const float* bb[3] = {(const float*)d_in[6],  (const float*)d_in[10], (const float*)d_in[14]};
    const float* wf1  = (const float*)d_in[15];
    const float* bf1v = (const float*)d_in[16];
    const float* wf2  = (const float*)d_in[17];
    const float* bf2v = (const float*)d_in[18];

    const int N = in_sizes[0] / DFEAT;
    const int E = in_sizes[1] / 2;
    const long long* srcIdx = ei;
    const long long* dstIdx = ei + E;

    // Workspace layout
    char* ws = (char*)d_ws;
    size_t nodeBytes = (size_t)N * DFEAT * sizeof(float);
    float*  bufA   = (float*)(ws);                       // fp32 activations
    float*  bufB   = (float*)(ws + nodeBytes);           // fp32 agg
    __bf16* bufC   = (__bf16*)(ws + 2 * nodeBytes);      // bf16 mid-layer activations
    __bf16* wbf    = (__bf16*)(ws + 2 * nodeBytes + nodeBytes / 2);   // 6 x 128x128 bf16
    float*  pooled = (float*)((char*)wbf + 6 * DFEAT * DFEAT * sizeof(__bf16));
    float*  counts = pooled + NGRAPH * DFEAT;

    // 1) Build fragment-swizzled bf16 weights (6 x 32KB; trivial).
    for (int l = 0; l < 3; ++l) {
        build_wfrag<<<(DFEAT * DFEAT + 255) / 256, 256, 0, stream>>>(wa[l], wbf + (2 * l) * DFEAT * DFEAT);
        build_wfrag<<<(DFEAT * DFEAT + 255) / 256, 256, 0, stream>>>(wb[l], wbf + (2 * l + 1) * DFEAT * DFEAT);
    }

    const int n4      = N * DFEAT / 4;
    const int blocksM = (N + 31) / 32;
    const float* h = x;
    for (int l = 0; l < 3; ++l) {
        // agg = h
        copy_f4<<<4096, 256, 0, stream>>>((const float4*)h, (float4*)bufB, n4);
        // agg += scatter(h[src] -> dst)
        scatter_add_edges<<<(E + 7) / 8, 256, 0, stream>>>(h, srcIdx, dstIdx, bufB, E);
        // two WMMA GEMMs with bias+relu (fp32 -> bf16 -> fp32)
        gemm_bias_relu<float, __bf16><<<blocksM, 256, 0, stream>>>(
                bufB, wbf + (2 * l) * DFEAT * DFEAT, ba[l], bufC, N);
        gemm_bias_relu<__bf16, float><<<blocksM, 256, 0, stream>>>(
                bufC, wbf + (2 * l + 1) * DFEAT * DFEAT, bb[l], bufA, N);
        h = bufA;
    }

    // mean-pool + FFN + softmax
    zero_f<<<(NGRAPH * DFEAT + NGRAPH + 255) / 256, 256, 0, stream>>>(pooled, NGRAPH * DFEAT + NGRAPH);
    pool_scatter<<<(N + 1) / 2, 256, 0, stream>>>(h, batch, pooled, counts, N);
    ffn_softmax<<<NGRAPH, 128, 0, stream>>>(pooled, counts, wf1, bf1v, wf2, bf2v, (float*)d_out);
}
